// WindowMSA_17652315586689
// MI455X (gfx1250) — matrix operand order repaired and checked by reading, not compiled, not run on hardware
//
#include <hip/hip_runtime.h>

typedef __attribute__((ext_vector_type(16))) _Float16 v16h;
typedef __attribute__((ext_vector_type(8)))  float    v8f;

#define WSZ   7
#define DSP   3
#define HEADS 6
#define HD    32
#define CDIM  192
#define IMG   56
#define NWH   8
#define NWW   8
#define NTOK  49
#define MPAD  64
#define BATCH 16
#define BWIN  (BATCH*64)      // 1024 windows
#define NQKV  576
#define NGRP  (BWIN*HEADS)    // 6144 (window,head) instances

union H2 { _Float16 h[2]; unsigned int u; };

__device__ __forceinline__ v8f wmma16(v16h a, v16h b, v8f c) {
  // D = A(16x32 f16) * B(32x16 f16) + C(16x16 f32)
  return __builtin_amdgcn_wmma_f32_16x16x32_f16(false, a, false, b, (short)0, c, false, false);
}

// A-operand element mapping (16x32 f16): k = (j>>3)*16 + (lane>>4)*8 + (j&7)
__device__ __forceinline__ void a_slot(int k, int m, int& lane, int& j) {
  lane = (((k >> 3) & 1) << 4) | m;
  j    = ((k >> 4) << 3) | (k & 7);
}

// ---- prep: f32 weight [K x rowStride] -> f16 B-operand tiles [nc][kk][lane][j]
__global__ __launch_bounds__(256) void k_prep_w(const float* __restrict__ w,
                                                _Float16* __restrict__ wsw,
                                                int nTotal, int rowStride) {
  int idx = blockIdx.x * 256 + threadIdx.x;
  if (idx >= nTotal) return;
  int j    = idx & 15;
  int lane = (idx >> 4) & 31;
  int t    = idx >> 9;            // nc*6 + kk
  int kk   = t % 6, nc = t / 6;
  int k = kk * 32 + (j >> 3) * 16 + (lane >> 4) * 8 + (j & 7);
  int n = nc * 16 + (lane & 15);
  wsw[idx] = (_Float16)w[(long)k * rowStride + n];
}

// ---- prep: bake pos-bias + shift masks + pad mask into 4 variants of [64][64]
__global__ __launch_bounds__(256) void k_prep_bias(const float* __restrict__ pos_emb,
                                                   float* __restrict__ tab) {
  int idx = blockIdx.x * 256 + threadIdx.x;       // 4 * 64 * 64
  if (idx >= 4 * 64 * 64) return;
  int col = idx & 63, row = (idx >> 6) & 63, v = idx >> 12;
  bool mUL = (v & 2) != 0, mLR = (v & 1) != 0;
  float add;
  if (col >= NTOK)      add = -1e9f;              // pad keys
  else if (row >= NTOK) add = 0.f;                // pad queries (discarded)
  else {
    int ih = row / WSZ, iw = row % WSZ, jh = col / WSZ, jw = col % WSZ;
    add = pos_emb[(jh - ih + 6) * 13 + (jw - iw + 6)];
    if (mUL && ((ih >= 4) != (jh >= 4))) add += -1e9f;
    if (mLR && ((iw >= 4) != (jw >= 4))) add += -1e9f;
  }
  tab[idx] = add;
}

// ---------------- K1: fused roll + QKV GEMM, outputs pre-swizzled Q/K/V -------
__global__ __launch_bounds__(128) void k_qkv(const float* __restrict__ x,
                                             const _Float16* __restrict__ wq_sw,
                                             _Float16* __restrict__ q_sw,
                                             _Float16* __restrict__ k_sw,
                                             _Float16* __restrict__ v_sw) {
  // A tile in A-operand order, u32-packed: [mtile][kk][lane][j/2]
  __shared__ __attribute__((aligned(32))) unsigned int aSw[4 * 6 * 32 * 8];
  int bwin = blockIdx.x;
  int b = bwin >> 6, win = bwin & 63;
  int nh = win >> 3, nwc = win & 7;
  int tid = threadIdx.x;

  for (int p = tid; p < MPAD * (CDIM / 4); p += 128) {
    int row = p / (CDIM / 4);
    int col = (p % (CDIM / 4)) * 4;
    float4 xv = make_float4(0.f, 0.f, 0.f, 0.f);
    if (row < NTOK) {
      int wh = row / WSZ, ww = row % WSZ;
      int sh = (nh * WSZ + wh + DSP) % IMG;       // roll(x, (-3,-3))
      int sw = (nwc * WSZ + ww + DSP) % IMG;
      xv = *(const float4*)(x + ((long)(b * IMG + sh) * IMG + sw) * CDIM + col);
    }
    int kk = col >> 5, kw = col & 31, lane2, j2;
    a_slot(kw, row & 15, lane2, j2);              // j2 multiple of 4 here
    H2 h0, h1;
    h0.h[0] = (_Float16)xv.x; h0.h[1] = (_Float16)xv.y;
    h1.h[0] = (_Float16)xv.z; h1.h[1] = (_Float16)xv.w;
    unsigned int* d = &aSw[(((row >> 4) * 6 + kk) * 32 + lane2) * 8 + (j2 >> 1)];
    d[0] = h0.u; d[1] = h1.u;
  }
  __syncthreads();

  int wave = tid >> 5, lane = tid & 31;
  int mC = lane & 15, khC = lane >> 4;
  int kh2 = (mC >> 3) & 1;
  const float scale = 0.17677669529663687f;       // HEAD_DIM^-0.5 folded into Q

  v16h aFrag[6];                                  // A fragments reused by all 36 chunks
  #pragma unroll
  for (int kk = 0; kk < 6; ++kk)
    aFrag[kk] = *(const v16h*)&aSw[((wave * 6 + kk) * 32 + lane) * 8];

  const _Float16* bp = wq_sw + (long)lane * 16;
  long gBase = (long)bwin * HEADS;

  // ---- Q chunks (nc = head*2 + dh): A-operand layout, stride-16 stores ----
  for (int head = 0; head < HEADS; ++head) {
    #pragma unroll
    for (int dh = 0; dh < 2; ++dh) {
      int nc = head * 2 + dh;
      v8f acc = {};
      #pragma unroll
      for (int kk = 0; kk < 6; ++kk)
        acc = wmma16(aFrag[kk], *(const v16h*)&bp[(long)(nc * 6 + kk) * 512], acc);
      _Float16* qp = q_sw + ((gBase + head) * 4 + wave) * 512
                   + (kh2 * 16 + khC * 8) * 16 + ((dh << 3) | (mC & 7));
      #pragma unroll
      for (int r = 0; r < 8; ++r) qp[r * 16] = (_Float16)(acc[r] * scale);
    }
  }
  // ---- K chunks (nc = 12 + head*2 + dh): B-operand layout, stride-16 stores ----
  for (int head = 0; head < HEADS; ++head) {
    #pragma unroll
    for (int dh = 0; dh < 2; ++dh) {
      int nc = 12 + head * 2 + dh;
      v8f acc = {};
      #pragma unroll
      for (int kk = 0; kk < 6; ++kk)
        acc = wmma16(aFrag[kk], *(const v16h*)&bp[(long)(nc * 6 + kk) * 512], acc);
      _Float16* kp = k_sw + ((gBase + head) * 4 + wave) * 512
                   + (kh2 * 16 + khC * 8) * 16 + ((dh << 3) | (mC & 7));
      #pragma unroll
      for (int r = 0; r < 8; ++r) kp[r * 16] = (_Float16)acc[r];
    }
  }
  // ---- V chunks (nc = 24 + head*2 + dh): lane-contiguous -> one b128 store ----
  for (int head = 0; head < HEADS; ++head) {
    #pragma unroll
    for (int dh = 0; dh < 2; ++dh) {
      int nc = 24 + head * 2 + dh;
      v8f acc = {};
      #pragma unroll
      for (int kk = 0; kk < 6; ++kk)
        acc = wmma16(aFrag[kk], *(const v16h*)&bp[(long)(nc * 6 + kk) * 512], acc);
      H2 p0, p1, p2, p3;
      p0.h[0] = (_Float16)acc[0]; p0.h[1] = (_Float16)acc[1];
      p1.h[0] = (_Float16)acc[2]; p1.h[1] = (_Float16)acc[3];
      p2.h[0] = (_Float16)acc[4]; p2.h[1] = (_Float16)acc[5];
      p3.h[0] = (_Float16)acc[6]; p3.h[1] = (_Float16)acc[7];
      _Float16* vp = v_sw + (((gBase + head) * 2 + (wave >> 1)) * 2 + dh) * 512
                   + (khC * 16 + mC) * 16 + (wave & 1) * 8;
      *(uint4*)vp = make_uint4(p0.u, p1.u, p2.u, p3.u);
    }
  }
}

// ---------------- K2: per-(window,head) attention -----------------------------
__global__ __launch_bounds__(64) void k_attn(const _Float16* __restrict__ q_sw,
                                             const _Float16* __restrict__ k_sw,
                                             const _Float16* __restrict__ v_sw,
                                             const float* __restrict__ biasTab,
                                             _Float16* __restrict__ o_sw) {
  __shared__ __attribute__((aligned(32))) _Float16 pSw[4 * 2 * 32 * 16]; // A-layout P
  int g = blockIdx.x;
  int bwin = g / HEADS, head = g % HEADS;
  int b = bwin >> 6, win = bwin & 63;
  int nh = win >> 3, nwc = win & 7;
  const float* tb = biasTab + (((nh == NWH - 1) ? 2 : 0) + ((nwc == NWW - 1) ? 1 : 0)) * 4096;
  int tid = threadIdx.x;
  int wave = tid >> 5, lane = tid & 31;
  int mC = lane & 15, khC = lane >> 4;
  const _Float16* qg = q_sw + (long)g * 2048;
  const _Float16* kg = k_sw + (long)g * 2048;
  const _Float16* vg = v_sw + (long)g * 2048;

  // ---- S = Q @ K^T ----
  v8f s[2][4];
  #pragma unroll
  for (int mt = 0; mt < 2; ++mt) {
    int mtile = wave * 2 + mt;
    v16h a = *(const v16h*)&qg[(mtile * 512) + lane * 16];
    #pragma unroll
    for (int nt = 0; nt < 4; ++nt) {
      v16h bf = *(const v16h*)&kg[(nt * 512) + lane * 16];
      v8f z = {};
      s[mt][nt] = wmma16(a, bf, z);
    }
  }

  // ---- bias(+masks) from table + softmax (f32) ----
  #pragma unroll
  for (int mt = 0; mt < 2; ++mt) {
    int mtile = wave * 2 + mt;
    #pragma unroll
    for (int r = 0; r < 8; ++r) {
      int row = mtile * 16 + khC * 8 + r;
      const float* trow = tb + row * 64 + mC;
      float vals[4];
      float mx = -3.0e38f;
      #pragma unroll
      for (int nt = 0; nt < 4; ++nt) {
        float sv = s[mt][nt][r] + trow[nt * 16];
        vals[nt] = sv;
        mx = fmaxf(mx, sv);
      }
      #pragma unroll
      for (int off = 1; off < 16; off <<= 1)
        mx = fmaxf(mx, __shfl_xor(mx, off, 32));
      float sum = 0.f;
      #pragma unroll
      for (int nt = 0; nt < 4; ++nt) {
        float e = __expf(vals[nt] - mx);
        vals[nt] = e;
        sum += e;
      }
      #pragma unroll
      for (int off = 1; off < 16; off <<= 1)
        sum += __shfl_xor(sum, off, 32);
      float is = 1.0f / sum;
      // store P directly in A-operand order: [mtile][ks][lane][j], k = col
      #pragma unroll
      for (int nt = 0; nt < 4; ++nt) {
        int col = nt * 16 + mC;
        int l2, j2; a_slot(col & 31, row & 15, l2, j2);
        pSw[((mtile * 2 + (col >> 5)) * 32 + l2) * 16 + j2] = (_Float16)(vals[nt] * is);
      }
    }
  }
  __syncthreads();

  // ---- O = P @ V ----
  v8f o[2][2] = {};
  #pragma unroll
  for (int mt = 0; mt < 2; ++mt) {
    int mtile = wave * 2 + mt;
    #pragma unroll
    for (int ks = 0; ks < 2; ++ks) {
      v16h a = *(const v16h*)&pSw[((mtile * 2 + ks) * 32 + lane) * 16];
      #pragma unroll
      for (int nt = 0; nt < 2; ++nt) {
        v16h bf = *(const v16h*)&vg[((ks * 2 + nt) * 512) + lane * 16];
        o[mt][nt] = wmma16(a, bf, o[mt][nt]);
      }
    }
  }

  // ---- scatter into K3's A-operand tiles: o_sw[tile][mtile][kk][lane][j] ----
  #pragma unroll
  for (int mt = 0; mt < 2; ++mt) {
    int mtile = wave * 2 + mt;
    #pragma unroll
    for (int r = 0; r < 8; ++r) {
      int row = mtile * 16 + khC * 8 + r;
      if (row < NTOK) {
        int wh = row / WSZ, ww = row % WSZ;
        int pix = ((b * IMG + nh * WSZ + wh) * IMG) + nwc * WSZ + ww;
        int tile = pix >> 6, r64 = pix & 63;
        #pragma unroll
        for (int nt = 0; nt < 2; ++nt) {
          int col = head * HD + nt * 16 + mC;     // global channel
          int kkO = col >> 5, kw = col & 31;
          int l2, j2; a_slot(kw, r64 & 15, l2, j2);
          o_sw[(((long)tile * 4 + (r64 >> 4)) * 6 + kkO) * 512 + l2 * 16 + j2] =
              (_Float16)o[mt][nt][r];
        }
      }
    }
  }
}

// ---------------- K3: output projection + bias + inverse roll (no LDS) --------
__global__ __launch_bounds__(128) void k_proj(const _Float16* __restrict__ o_sw,
                                              const _Float16* __restrict__ wo_sw,
                                              const float* __restrict__ b_out,
                                              float* __restrict__ out) {
  int tile = blockIdx.x;
  int tid = threadIdx.x;
  int wave = tid >> 5, lane = tid & 31;
  int mC = lane & 15, khC = lane >> 4;

  const _Float16* ap = o_sw + ((long)tile * 4 + wave) * 6 * 512;
  v16h aFrag[6];
  #pragma unroll
  for (int kk = 0; kk < 6; ++kk)
    aFrag[kk] = *(const v16h*)&ap[kk * 512 + lane * 16];

  // hoisted roll addressing (depends only on r)
  long obase[8];
  #pragma unroll
  for (int r = 0; r < 8; ++r) {
    int grow = tile * MPAD + wave * 16 + khC * 8 + r;
    int bb2 = grow / (IMG * IMG);
    int rem = grow % (IMG * IMG);
    int sh = rem / IMG, sw = rem % IMG;
    int dh = sh + DSP; if (dh >= IMG) dh -= IMG;  // roll(out, (+3,+3))
    int dw = sw + DSP; if (dw >= IMG) dw -= IMG;
    obase[r] = ((long)(bb2 * IMG + dh) * IMG + dw) * CDIM;
  }

  for (int nc = 0; nc < CDIM / 16; ++nc) {
    v8f acc = {};
    #pragma unroll
    for (int kk = 0; kk < 6; ++kk) {
      v16h bf = *(const v16h*)&wo_sw[(long)((nc * 6 + kk) * 32 + lane) * 16];
      acc = wmma16(aFrag[kk], bf, acc);
    }
    int n = nc * 16 + mC;
    float bias = b_out[n];
    #pragma unroll
    for (int r = 0; r < 8; ++r)
      out[obase[r] + n] = acc[r] + bias;
  }
}

extern "C" void kernel_launch(void* const* d_in, const int* in_sizes, int n_in,
                              void* d_out, int out_size, void* d_ws, size_t ws_size,
                              hipStream_t stream) {
  const float* x     = (const float*)d_in[0];
  const float* w_qkv = (const float*)d_in[1];
  const float* pos   = (const float*)d_in[2];
  const float* w_out = (const float*)d_in[3];
  const float* b_out = (const float*)d_in[4];
  float* out = (float*)d_out;

  // workspace carve-up (all offsets 32B aligned)
  size_t perG = (size_t)NGRP * 2048;                 // halfs per q/k/v operand buffer
  _Float16* q_sw  = (_Float16*)d_ws;
  _Float16* k_sw  = q_sw + perG;
  _Float16* v_sw  = k_sw + perG;
  _Float16* o_sw  = v_sw + perG;                     // 784*4*6*512 halfs
  _Float16* wq_sw = o_sw + (size_t)784 * 4 * 6 * 512;
  _Float16* wo_sw = wq_sw + 36 * 6 * 512;
  float*    biasT = (float*)(wo_sw + 12 * 6 * 512);  // 4*64*64 f32

  const int nWQ = 36 * 6 * 512, nWO = 12 * 6 * 512;
  k_prep_w   <<<dim3((nWQ + 255) / 256), dim3(256), 0, stream>>>(w_qkv, wq_sw, nWQ, NQKV);
  k_prep_w   <<<dim3((nWO + 255) / 256), dim3(256), 0, stream>>>(w_out, wo_sw, nWO, CDIM);
  k_prep_bias<<<dim3(64),               dim3(256), 0, stream>>>(pos, biasT);

  k_qkv <<<dim3(BWIN), dim3(128), 0, stream>>>(x, wq_sw, q_sw, k_sw, v_sw);
  k_attn<<<dim3(NGRP), dim3(64),  0, stream>>>(q_sw, k_sw, v_sw, biasT, o_sw);
  k_proj<<<dim3((BATCH*IMG*IMG)/MPAD), dim3(128), 0, stream>>>(o_sw, wo_sw, b_out, out);
}
